// BClassifier_37546604101932
// MI455X (gfx1250) — compile-verified, matrix operand order checked by reading, not run on hardware
//
#include <hip/hip_runtime.h>
#include <math.h>
#include <float.h>

// ---------------- types ----------------
typedef __attribute__((ext_vector_type(16))) __bf16          v16bf;
typedef __attribute__((ext_vector_type(8)))  float           v8f;
typedef __attribute__((ext_vector_type(8)))  unsigned short  ushort8;

union BF16x16 { ushort8 h[2]; v16bf v; };

// ---------------- constants ----------------
#define NINST   50000
#define DSF     512
#define DDF     1024
#define KSEL    20
#define DATTF   256
#define HIDF    128
#define NB_RED  200      // blocks for softmax reductions

// output layout (flat, reference tuple order)
#define OFF_C     0
#define OFF_CD    1
#define OFF_AD    2
#define OFF_B     50002
#define OFF_BD    50022
#define OFF_AAUX  100022

// input leaf indices (JAX pytree: sorted dict keys)
#define I_FEATS   0
#define I_FDEEP   1
#define I_AD_AW   2
#define I_AD_AB   3
#define I_AD_BW   4
#define I_AD_BB   5
#define I_AD_CW   6
#define I_AD_CB   7
#define I_AX_AW   8
#define I_AX_AB   9
#define I_AX_BW   10
#define I_AX_BB   11
#define I_AX_CW   12
#define I_AX_CB   13
#define I_CLS_W   14
#define I_CLS_B   15
#define I_CLSD_W  16
#define I_CLSD_B  17
#define I_MIX0    18   // per layer: ch1W,ch1b,ch2W,ch2b,tok1W,tok1b,tok2W,tok2b
#define I_PROJ_W  34
#define I_PROJ_B  35

struct Ptrs { const float* p[36]; };

// ---------------- helpers ----------------
__device__ __forceinline__ unsigned short f2bf(float f) {
    unsigned int u = __float_as_uint(f);
    unsigned int r = u + 0x7FFFu + ((u >> 16) & 1u);   // RNE
    return (unsigned short)(r >> 16);
}
__device__ __forceinline__ float sigm(float x)  { return 1.f / (1.f + expf(-x)); }
__device__ __forceinline__ float gelu_exact(float x) {
    return 0.5f * x * (1.f + erff(x * 0.70710678118654752440f));
}

// ---------------- kernel 0: pack f32 weights -> bf16 WMMA B-operand layout ---
// dest block (kt,nt) = 32 lanes x 16 bf16 contiguous (1024B).
// lane l<16 : col = nt*16+l,      K = {0..7}(e0..7), {16..23}(e8..15)
// lane l>=16: col = nt*16+(l-16), K = {8..15},       {24..31}
__global__ void pack_kernel(const float* __restrict__ W, unsigned short* __restrict__ dst,
                            int fin, int fout) {
    long total = (long)fin * fout;
    int ntiles = fout >> 4;
    for (long d = (long)blockIdx.x * blockDim.x + threadIdx.x; d < total;
         d += (long)gridDim.x * blockDim.x) {
        int e  = (int)(d & 15);
        int l  = (int)((d >> 4) & 31);
        long blk = d >> 9;
        int nt = (int)(blk % ntiles);
        int kt = (int)(blk / ntiles);
        int kl = (l < 16) ? ((e < 8) ? e : e + 8) : ((e < 8) ? e + 8 : e + 16);
        int k  = kt * 32 + kl;
        int n  = nt * 16 + (l & 15);
        dst[d] = f2bf(W[(long)k * fout + n]);
    }
}

// ---------------- kernel 1: fused deep path (WMMA bf16) ----------------
// one wave (32 lanes) per block, 16 rows of feats_deep per block.
// Produces per-row: attention logit and s[i] = relu-V[i] @ Wcls_deep.
__global__ __launch_bounds__(32)
void deep_kernel(Ptrs P, const unsigned short* __restrict__ wpPk,
                 const unsigned short* __restrict__ waPk,
                 const unsigned short* __restrict__ wbPk,
                 float* __restrict__ logit, float* __restrict__ sdot) {
    __shared__ unsigned short Xl[16 * 1024];   // 32 KB, bf16 row tile of feats_deep
    __shared__ unsigned short Vl[16 * 1024];   // 32 KB, bf16 V tile (post-relu)
    __shared__ float redC[16][16];
    __shared__ float redS[16][16];

    const int lane = threadIdx.x;
    const int nl   = lane & 15;
    const int hi   = lane >> 4;
    const long rowbase = (long)blockIdx.x * 16;

    // stage 16x1024 f32 -> bf16 LDS (vectorized b128 loads)
    const float4* src4 = (const float4*)(P.p[I_FDEEP] + rowbase * DDF);
    for (int t = lane; t < 16 * 1024 / 4; t += 32) {
        float4 f = src4[t];
        unsigned int w0 = (unsigned int)f2bf(f.x) | ((unsigned int)f2bf(f.y) << 16);
        unsigned int w1 = (unsigned int)f2bf(f.z) | ((unsigned int)f2bf(f.w) << 16);
        ((unsigned int*)Xl)[t * 2 + 0] = w0;
        ((unsigned int*)Xl)[t * 2 + 1] = w1;
    }
    __syncthreads();

    const float* bproj = P.p[I_PROJ_B];
    const float* wclsd = P.p[I_CLSD_W];

    float sacc[8];
#pragma unroll
    for (int r = 0; r < 8; r++) sacc[r] = 0.f;

    // GEMM1: V = relu(X @ Wproj + b), streamed by 16-column tiles
    for (int ct = 0; ct < 64; ++ct) {
        float bv = bproj[ct * 16 + nl];
        v8f acc;
#pragma unroll
        for (int r = 0; r < 8; r++) acc[r] = bv;
        for (int kt = 0; kt < 32; ++kt) {
            BF16x16 A, B;
            int abase = kt * 32 + hi * 8;
            A.h[0] = *(const ushort8*)&Xl[nl * 1024 + abase];
            A.h[1] = *(const ushort8*)&Xl[nl * 1024 + abase + 16];
            const ushort8* wb = (const ushort8*)&wpPk[(size_t)(kt * 64 + ct) * 512 + lane * 16];
            B.h[0] = wb[0];
            B.h[1] = wb[1];
            acc = __builtin_amdgcn_wmma_f32_16x16x32_bf16(false, A.v, false, B.v,
                                                          (short)0, acc, false, false);
        }
        float wc = wclsd[ct * 16 + nl];
#pragma unroll
        for (int r = 0; r < 8; r++) {
            float v = acc[r];
            v = v > 0.f ? v : 0.f;                       // relu
            sacc[r] += v * wc;                           // fold V @ Wcls_deep
            Vl[(r + hi * 8) * 1024 + ct * 16 + nl] = f2bf(v);
        }
    }
    __syncthreads();

    // GEMM2: gated attention  c = (tanh(V@Wa+ba) * sigmoid(V@Wb+bb)) @ Wc
    const float* ba_ = P.p[I_AD_AB];
    const float* bb_ = P.p[I_AD_BB];
    const float* wcA = P.p[I_AD_CW];
    float cacc[8];
#pragma unroll
    for (int r = 0; r < 8; r++) cacc[r] = 0.f;

    for (int jt = 0; jt < 16; ++jt) {
        float bav = ba_[jt * 16 + nl];
        float bbv = bb_[jt * 16 + nl];
        float wcv = wcA[jt * 16 + nl];
        v8f accA, accB;
#pragma unroll
        for (int r = 0; r < 8; r++) { accA[r] = bav; accB[r] = bbv; }
        for (int kt = 0; kt < 32; ++kt) {
            BF16x16 A, Ba, Bb;
            int abase = kt * 32 + hi * 8;
            A.h[0] = *(const ushort8*)&Vl[nl * 1024 + abase];
            A.h[1] = *(const ushort8*)&Vl[nl * 1024 + abase + 16];
            const ushort8* wa  = (const ushort8*)&waPk[(size_t)(kt * 16 + jt) * 512 + lane * 16];
            const ushort8* wbp = (const ushort8*)&wbPk[(size_t)(kt * 16 + jt) * 512 + lane * 16];
            Ba.h[0] = wa[0];  Ba.h[1] = wa[1];
            Bb.h[0] = wbp[0]; Bb.h[1] = wbp[1];
            accA = __builtin_amdgcn_wmma_f32_16x16x32_bf16(false, A.v, false, Ba.v,
                                                           (short)0, accA, false, false);
            accB = __builtin_amdgcn_wmma_f32_16x16x32_bf16(false, A.v, false, Bb.v,
                                                           (short)0, accB, false, false);
        }
#pragma unroll
        for (int r = 0; r < 8; r++) {
            float h = tanhf(accA[r]) * sigm(accB[r]);
            cacc[r] += h * wcv;
        }
    }

    // lane reduction (16 N-lanes per row) via LDS
#pragma unroll
    for (int r = 0; r < 8; r++) {
        redC[r + hi * 8][nl] = cacc[r];
        redS[r + hi * 8][nl] = sacc[r];
    }
    __syncthreads();
    if (lane < 16) {
        float sc = 0.f, ss = 0.f;
        for (int j = 0; j < 16; j++) { sc += redC[lane][j]; ss += redS[lane][j]; }
        logit[rowbase + lane] = sc + P.p[I_AD_CB][0];
        sdot[rowbase + lane]  = ss;
    }
}

// ---------------- softmax stats (deterministic tree reductions) ----------------
__global__ void stats_max1(const float* __restrict__ logit, float* __restrict__ bmax) {
    __shared__ float red[256];
    float m = -FLT_MAX;
    for (int i = blockIdx.x * blockDim.x + threadIdx.x; i < NINST;
         i += gridDim.x * blockDim.x)
        m = fmaxf(m, logit[i]);
    red[threadIdx.x] = m;
    __syncthreads();
    for (int s = 128; s > 0; s >>= 1) {
        if (threadIdx.x < s) red[threadIdx.x] = fmaxf(red[threadIdx.x], red[threadIdx.x + s]);
        __syncthreads();
    }
    if (threadIdx.x == 0) bmax[blockIdx.x] = red[0];
}

__global__ void stats_max2(const float* __restrict__ bmax, float* __restrict__ sc) {
    __shared__ float red[256];
    red[threadIdx.x] = (threadIdx.x < NB_RED) ? bmax[threadIdx.x] : -FLT_MAX;
    __syncthreads();
    for (int s = 128; s > 0; s >>= 1) {
        if (threadIdx.x < s) red[threadIdx.x] = fmaxf(red[threadIdx.x], red[threadIdx.x + s]);
        __syncthreads();
    }
    if (threadIdx.x == 0) sc[0] = red[0];
}

__global__ void stats_sum1(const float* __restrict__ logit, const float* __restrict__ sdot,
                           const float* __restrict__ sc, float* __restrict__ bsum,
                           float* __restrict__ bdot) {
    __shared__ float rs[256];
    __shared__ float rd[256];
    float gmax = sc[0];
    float se = 0.f, sd = 0.f;
    for (int i = blockIdx.x * blockDim.x + threadIdx.x; i < NINST;
         i += gridDim.x * blockDim.x) {
        float e = expf(logit[i] - gmax);
        se += e;
        sd += e * sdot[i];
    }
    rs[threadIdx.x] = se; rd[threadIdx.x] = sd;
    __syncthreads();
    for (int s = 128; s > 0; s >>= 1) {
        if (threadIdx.x < s) { rs[threadIdx.x] += rs[threadIdx.x + s];
                               rd[threadIdx.x] += rd[threadIdx.x + s]; }
        __syncthreads();
    }
    if (threadIdx.x == 0) { bsum[blockIdx.x] = rs[0]; bdot[blockIdx.x] = rd[0]; }
}

__global__ void stats_sum2(const float* __restrict__ bsum, const float* __restrict__ bdot,
                           float* __restrict__ sc, const float* __restrict__ bclsd,
                           float* __restrict__ out) {
    __shared__ float rs[256];
    __shared__ float rd[256];
    rs[threadIdx.x] = (threadIdx.x < NB_RED) ? bsum[threadIdx.x] : 0.f;
    rd[threadIdx.x] = (threadIdx.x < NB_RED) ? bdot[threadIdx.x] : 0.f;
    __syncthreads();
    for (int s = 128; s > 0; s >>= 1) {
        if (threadIdx.x < s) { rs[threadIdx.x] += rs[threadIdx.x + s];
                               rd[threadIdx.x] += rd[threadIdx.x + s]; }
        __syncthreads();
    }
    if (threadIdx.x == 0) {
        sc[1] = rs[0];
        sc[2] = rd[0];
        // sum(B_deep) = sum(A_deep*s) + N*b = gdot/gsum + N*b
        out[OFF_CD] = sigm(rd[0] / rs[0] + (float)NINST * bclsd[0]);
    }
}

__global__ void norm_kernel(const float* __restrict__ logit, const float* __restrict__ sdot,
                            const float* __restrict__ sc, const float* __restrict__ bclsd,
                            float* __restrict__ out) {
    float gmax = sc[0];
    float inv  = 1.f / sc[1];
    float b    = bclsd[0];
    for (int i = blockIdx.x * blockDim.x + threadIdx.x; i < NINST;
         i += gridDim.x * blockDim.x) {
        float a = expf(logit[i] - gmax) * inv;
        out[OFF_AD + i] = a;
        out[OFF_BD + i] = a * sdot[i] + b;
    }
}

// ---------------- top-K (stable: ties -> lowest index, as argsort) ------------
__global__ __launch_bounds__(1024)
void topk_kernel(const float* __restrict__ logit, int* __restrict__ topk) {
    __shared__ float sV[1024];
    __shared__ int   sI[1024];
    __shared__ int   sel[KSEL];
    const int tid = threadIdx.x;
    for (int it = 0; it < KSEL; ++it) {
        float best = -FLT_MAX;
        int bidx = 0x7fffffff;
        for (int i = tid; i < NINST; i += 1024) {
            bool skip = false;
            for (int j = 0; j < it; ++j) skip = skip || (sel[j] == i);
            if (skip) continue;
            float v = logit[i];
            if (v > best || (v == best && i < bidx)) { best = v; bidx = i; }
        }
        sV[tid] = best; sI[tid] = bidx;
        __syncthreads();
        for (int s = 512; s > 0; s >>= 1) {
            if (tid < s) {
                float v2 = sV[tid + s]; int i2 = sI[tid + s];
                if (v2 > sV[tid] || (v2 == sV[tid] && i2 < sI[tid])) { sV[tid] = v2; sI[tid] = i2; }
            }
            __syncthreads();
        }
        if (tid == 0) { sel[it] = sI[0]; topk[it] = sI[0]; }
        __syncthreads();
    }
}

// ---------------- mixer + aux attention + quantile + heads (single block) -----
__global__ __launch_bounds__(256)
void mixer_kernel(Ptrs P, const int* __restrict__ topk, float* __restrict__ out) {
    __shared__ float X[DSF * KSEL];    // x[d][k], 40 KB
    __shared__ float T1[KSEL * HIDF];  // 10 KB
    __shared__ float AA[DSF];
    __shared__ float SORT[DSF];
    __shared__ float red[256];
    const int tid = threadIdx.x;
    const float* feats = P.p[I_FEATS];

    // gather: x = topk_feats.T  -> X[d*20+k]
    for (int t = tid; t < DSF * KSEL; t += 256) {
        int d = t / KSEL, k = t % KSEL;
        X[t] = feats[(long)topk[k] * DSF + d];
    }
    __syncthreads();

    for (int layer = 0; layer < 4; ++layer) {
        const float* ch1W  = P.p[I_MIX0 + layer * 8 + 0];
        const float* ch1b  = P.p[I_MIX0 + layer * 8 + 1];
        const float* ch2W  = P.p[I_MIX0 + layer * 8 + 2];
        const float* ch2b  = P.p[I_MIX0 + layer * 8 + 3];
        const float* tok1W = P.p[I_MIX0 + layer * 8 + 4];
        const float* tok1b = P.p[I_MIX0 + layer * 8 + 5];
        const float* tok2W = P.p[I_MIX0 + layer * 8 + 6];
        const float* tok2b = P.p[I_MIX0 + layer * 8 + 7];

        // token mixing: T1[k][j] = gelu(sum_d X[d][k]*W1[d][j] + b1[j])
        for (int t = tid; t < KSEL * HIDF; t += 256) {
            int k = t / HIDF, j = t % HIDF;
            float acc = tok1b[j];
            for (int d = 0; d < DSF; ++d) acc += X[d * KSEL + k] * tok1W[d * HIDF + j];
            T1[t] = gelu_exact(acc);
        }
        __syncthreads();
        // X[d][k] += sum_j T1[k][j]*W2[j][d] + b2[d]
        for (int t = tid; t < DSF * KSEL; t += 256) {
            int d = t / KSEL, k = t % KSEL;
            float acc = tok2b[d];
            for (int j = 0; j < HIDF; ++j) acc += T1[k * HIDF + j] * tok2W[j * DSF + d];
            X[t] += acc;
        }
        __syncthreads();

        // channel mixing (row-local, 2 rows per thread)
        for (int d = tid; d < DSF; d += 256) {
            float xl[KSEL], accs[KSEL];
#pragma unroll
            for (int k = 0; k < KSEL; ++k) { xl[k] = X[d * KSEL + k]; accs[k] = ch2b[k]; }
            for (int j = 0; j < HIDF; ++j) {
                float h = ch1b[j];
#pragma unroll
                for (int k = 0; k < KSEL; ++k) h += xl[k] * ch1W[k * HIDF + j];
                h = gelu_exact(h);
#pragma unroll
                for (int k = 0; k < KSEL; ++k) accs[k] += h * ch2W[j * KSEL + k];
            }
#pragma unroll
            for (int k = 0; k < KSEL; ++k) X[d * KSEL + k] = xl[k] + accs[k];
        }
        __syncthreads();
    }

    // aux gated attention: AA[d] = sum_j tanh(x@Wa+ba)*sigm(x@Wb+bb)*Wc[j] + cb
    {
        const float* aW = P.p[I_AX_AW]; const float* ab = P.p[I_AX_AB];
        const float* bW = P.p[I_AX_BW]; const float* bb = P.p[I_AX_BB];
        const float* cW = P.p[I_AX_CW]; const float* cb = P.p[I_AX_CB];
        for (int d = tid; d < DSF; d += 256) {
            float s = cb[0];
            for (int j = 0; j < HIDF; ++j) {
                float va = ab[j], vb = bb[j];
#pragma unroll
                for (int k = 0; k < KSEL; ++k) {
                    float xv = X[d * KSEL + k];
                    va += xv * aW[k * HIDF + j];
                    vb += xv * bW[k * HIDF + j];
                }
                s += tanhf(va) * sigm(vb) * cW[j];
            }
            AA[d] = s;
        }
    }
    __syncthreads();

    // quantile(0.75) via bitonic sort (ascending) of 512 values
    for (int t = tid; t < DSF; t += 256) SORT[t] = AA[t];
    __syncthreads();
    for (int k = 2; k <= DSF; k <<= 1) {
        for (int j = k >> 1; j > 0; j >>= 1) {
            for (int t = tid; t < DSF; t += 256) {
                int ixj = t ^ j;
                if (ixj > t) {
                    bool up = ((t & k) == 0);
                    float a = SORT[t], b = SORT[ixj];
                    if ((a > b) == up) { SORT[t] = b; SORT[ixj] = a; }
                }
            }
            __syncthreads();
        }
    }
    // linear-interp quantile at 0.75*(512-1)=383.25
    float q = SORT[383] + 0.25f * (SORT[384] - SORT[383]);

    // std (ddof=1)
    float part = 0.f;
    for (int d = tid; d < DSF; d += 256) part += AA[d];
    red[tid] = part; __syncthreads();
    for (int s = 128; s > 0; s >>= 1) {
        if (tid < s) red[tid] += red[tid + s];
        __syncthreads();
    }
    float mean = red[0] / (float)DSF;
    __syncthreads();
    part = 0.f;
    for (int d = tid; d < DSF; d += 256) { float dv = AA[d] - mean; part += dv * dv; }
    red[tid] = part; __syncthreads();
    for (int s = 128; s > 0; s >>= 1) {
        if (tid < s) red[tid] += red[tid + s];
        __syncthreads();
    }
    float sdev = sqrtf(red[0] / (float)(DSF - 1));
    __syncthreads();

    // A_aux = sigmoid((AA - q)/(s+1e-6)*3)  (reuse SORT as final A_aux buffer)
    for (int d = tid; d < DSF; d += 256) {
        float v = sigm((AA[d] - q) / (sdev + 1e-6f) * 3.0f);
        SORT[d] = v;
        out[OFF_AAUX + d] = v;
    }
    __syncthreads();

    // B[k] = sum_d topk_feats[k][d]*A_aux[d]*Wcls[d] + bcls ; C = sigm(sum_k B)
    const float* clsW = P.p[I_CLS_W];
    float bacc = 0.f;
    if (tid < KSEL) {
        float acc = P.p[I_CLS_B][0];
        const float* frow = feats + (long)topk[tid] * DSF;
        for (int d = 0; d < DSF; ++d) acc += frow[d] * SORT[d] * clsW[d];
        out[OFF_B + tid] = acc;
        bacc = acc;
    }
    red[tid] = bacc; __syncthreads();
    for (int s = 128; s > 0; s >>= 1) {
        if (tid < s) red[tid] += red[tid + s];
        __syncthreads();
    }
    if (tid == 0) out[OFF_C] = sigm(red[0]);
}

// ---------------- launcher ----------------
extern "C" void kernel_launch(void* const* d_in, const int* in_sizes, int n_in,
                              void* d_out, int out_size, void* d_ws, size_t ws_size,
                              hipStream_t stream) {
    Ptrs P;
    for (int i = 0; i < 36; ++i) P.p[i] = (i < n_in) ? (const float*)d_in[i] : nullptr;
    float* out = (float*)d_out;

    char* ws = (char*)d_ws;
    unsigned short* wpPk = (unsigned short*)(ws + 0);         // 1024x1024 bf16: 2 MB
    unsigned short* waPk = (unsigned short*)(ws + 2097152);   // 1024x256 bf16
    unsigned short* wbPk = (unsigned short*)(ws + 2621440);   // 1024x256 bf16
    float* logit    = (float*)(ws + 3145728);                 // 50000 f32
    float* sdot     = (float*)(ws + 3345920);                 // 50000 f32
    float* bmax     = (float*)(ws + 3546112);
    float* bsum     = (float*)(ws + 3547136);
    float* bdot     = (float*)(ws + 3548160);
    float* scal     = (float*)(ws + 3549184);                 // gmax, gsum, gdot
    int*   topkIdx  = (int*)  (ws + 3549440);

    // pack weights to WMMA-friendly bf16
    pack_kernel<<<512, 256, 0, stream>>>(P.p[I_PROJ_W], wpPk, DDF, DDF);
    pack_kernel<<<256, 256, 0, stream>>>(P.p[I_AD_AW], waPk, DDF, DATTF);
    pack_kernel<<<256, 256, 0, stream>>>(P.p[I_AD_BW], wbPk, DDF, DATTF);

    // fused deep path: 3125 blocks x 1 wave x 16 rows
    deep_kernel<<<NINST / 16, 32, 0, stream>>>(P, wpPk, waPk, wbPk, logit, sdot);

    // softmax stats + normalization + B_deep / C_deep
    stats_max1<<<NB_RED, 256, 0, stream>>>(logit, bmax);
    stats_max2<<<1, 256, 0, stream>>>(bmax, scal);
    stats_sum1<<<NB_RED, 256, 0, stream>>>(logit, sdot, scal, bsum, bdot);
    stats_sum2<<<1, 256, 0, stream>>>(bsum, bdot, scal, P.p[I_CLSD_B], out);
    norm_kernel<<<NB_RED, 256, 0, stream>>>(logit, sdot, scal, P.p[I_CLSD_B], out);

    // top-20 selection (stable)
    topk_kernel<<<1, 1024, 0, stream>>>(logit, topkIdx);

    // mixer + aux attention + quantile + final heads
    mixer_kernel<<<1, 256, 0, stream>>>(P, topkIdx, out);
}
